// OverlappingCrossAttention_40097814676095
// MI455X (gfx1250) — matrix operand order repaired and checked by reading, not compiled, not run on hardware
//
#include <hip/hip_runtime.h>

// ---------------------------------------------------------------------------
// CDNA5 (gfx1250) implementation of overlapping-window cross attention.
// Matmuls: v_wmma_f32_16x16x32_f16 (wave32). K/V/Q window staging: TDM
// (tensor_load_to_lds) with LDS padding, guarded with a fallback path.
// ---------------------------------------------------------------------------

typedef __attribute__((ext_vector_type(16))) _Float16 v16h;
typedef __attribute__((ext_vector_type(8)))  float    v8f;
typedef __attribute__((ext_vector_type(4)))  unsigned v4u;
typedef __attribute__((ext_vector_type(4)))  int      v4i;
typedef __attribute__((ext_vector_type(8)))  int      v8i_t;

#define NUM_HEADS 8
#define E_DIM 32
#define MO 12
#define HP 68           // 64 + 2*PAD
#define NKEY 144        // MO*MO
#define NKEY_PAD 160    // padded to multiple of 32 for A@V k-loop

#if defined(__gfx1250__) && __has_builtin(__builtin_amdgcn_tensor_load_to_lds) && \
    __has_builtin(__builtin_amdgcn_s_wait_tensorcnt)
#define USE_TDM 1
#else
#define USE_TDM 0
#endif

static __device__ inline v8f zero8() {
  v8f z = {0.f,0.f,0.f,0.f,0.f,0.f,0.f,0.f};
  return z;
}

// A fragment: 16x32 f16, element (m,k) = src[(row0+m)*ld + k0 + k]
static __device__ inline v16h load_a_frag(const _Float16* src, int ld, int row0,
                                          int k0, int lane) {
  int m = lane & 15, half = lane >> 4;
  v16h a;
#pragma unroll
  for (int v = 0; v < 8; ++v) {
    int kb = half * 16 + 2 * v + k0;
    const _Float16* p = src + (row0 + m) * ld + kb;
    a[2 * v]     = p[0];
    a[2 * v + 1] = p[1];
  }
  return a;
}

// B fragment: 32x16 (KxN), element (k,n) stored as src[(n0+n)*ld + k0 + k]
static __device__ inline v16h load_b_frag_nk(const _Float16* src, int ld, int n0,
                                             int k0, int lane) {
  int n = lane & 15, half = lane >> 4;
  v16h b;
#pragma unroll
  for (int v = 0; v < 8; ++v) {
    int k = half * 16 + 2 * v + k0;
    const _Float16* p = src + (n0 + n) * ld + k;
    b[2 * v]     = p[0];
    b[2 * v + 1] = p[1];
  }
  return b;
}

// B fragment: 32x16 (KxN), element (k,n) stored as src[(k0+k)*ld + n0 + n]
static __device__ inline v16h load_b_frag_kn(const _Float16* src, int ld, int k0,
                                             int n0, int lane) {
  int n = lane & 15, half = lane >> 4;
  v16h b;
#pragma unroll
  for (int v = 0; v < 8; ++v) {
    int k = k0 + half * 16 + 2 * v;
    b[2 * v]     = src[k * ld + n0 + n];
    b[2 * v + 1] = src[(k + 1) * ld + n0 + n];
  }
  return b;
}

static __device__ inline v8f wmma_f16(v16h a, v16h b, v8f c) {
  return __builtin_amdgcn_wmma_f32_16x16x32_f16(false, a, false, b, (short)0, c,
                                                false, false);
}

#if USE_TDM
// LDS byte offset: for generic pointers into the LDS aperture, addr[31:0] IS
// the LDS offset (ISA 10.2 per-aperture calc).
static __device__ inline unsigned lds_off(const void* p) {
  return (unsigned)(unsigned long long)p;
}

// 2D TDM load of f16 data with LDS row padding of +1 DWORD every 16 DWORDs
// (pad_interval=3 -> 16 DWORDs, pad_amount=0 -> 1 DWORD). Converts each 32-f16
// logical row into a 34-f16 LDS row (bank-conflict-free fragment loads).
// Dims in elements (data_size = 2 bytes).
static __device__ inline void tdm_load_2d(unsigned ldsaddr, unsigned long long ga,
                                          unsigned tensor_d0, unsigned tensor_d1,
                                          unsigned tile_d0, unsigned tile_d1,
                                          unsigned stride0) {
  v4u g0;
  g0.x = 1u;                                           // count=1, user mode
  g0.y = ldsaddr;                                      // lds_addr [63:32]
  g0.z = (unsigned)(ga & 0xFFFFFFFFu);                 // global_addr[31:0]
  g0.w = (unsigned)((ga >> 32) & 0x01FFFFFFu) | (2u << 30);  // ga[56:32], type=2
  v8i_t g1;
  g1[0] = (int)((1u << 16) | (1u << 20) | (3u << 22)); // data_size=2B, pad_en, intv=16dw
  g1[1] = (int)(tensor_d0 << 16);                      // tensor_dim0[15:0] @ [63:48]
  g1[2] = (int)((tensor_d0 >> 16) | (tensor_d1 << 16));
  g1[3] = (int)((tensor_d1 >> 16) | (tile_d0 << 16));  // tile_dim0 @ [127:112]
  g1[4] = (int)(tile_d1 & 0xFFFFu);                    // tile_dim1, tile_dim2=0
  g1[5] = (int)stride0;                                // tensor_dim0_stride[31:0]
  g1[6] = 0;
  g1[7] = 0;
  v4i z4 = {0, 0, 0, 0};
#if __clang_major__ >= 23
  v8i_t z8 = {0, 0, 0, 0, 0, 0, 0, 0};
  __builtin_amdgcn_tensor_load_to_lds(g0, g1, z4, z4, z8, 0);
#else
  __builtin_amdgcn_tensor_load_to_lds(g0, g1, z4, z4, 0);
#endif
}
#endif  // USE_TDM

// ---------------------------------------------------------------------------
// 0) Convert W1 (768x256) and W2 (256x256) to f16
// ---------------------------------------------------------------------------
__global__ void k_prep(const float* __restrict__ W1, const float* __restrict__ W2,
                       _Float16* __restrict__ w1h, _Float16* __restrict__ w2h) {
  int idx = blockIdx.x * 256 + threadIdx.x;   // 262144 threads total
  if (idx < 196608) w1h[idx] = (_Float16)W1[idx];
  else              w2h[idx - 196608] = (_Float16)W2[idx - 196608];
}

// ---------------------------------------------------------------------------
// 1) Transpose x (B,C,H,W) f32 -> xt (B*H*W, C) f16, LDS-tiled
// ---------------------------------------------------------------------------
__global__ void k_transpose(const float* __restrict__ x, _Float16* __restrict__ xt) {
  __shared__ float tile[32][33];
  int b   = blockIdx.z;
  int hw0 = blockIdx.x * 32;
  int c0  = blockIdx.y * 32;
  int tx = threadIdx.x, ty = threadIdx.y;   // 32 x 8
#pragma unroll
  for (int i = 0; i < 4; ++i) {
    int c = c0 + ty + 8 * i;
    tile[ty + 8 * i][tx] = x[((size_t)(b * 256 + c)) * 4096 + hw0 + tx];
  }
  __syncthreads();
#pragma unroll
  for (int i = 0; i < 4; ++i) {
    int hw = hw0 + ty + 8 * i;
    xt[((size_t)(b * 4096 + hw)) * 256 + c0 + tx] = (_Float16)tile[tx][ty + 8 * i];
  }
}

// ---------------------------------------------------------------------------
// 2) QKV projection: qkv = xt @ W1^T + b1 (M=65536, N=768, K=256).
//    Epilogue de-interleaves (col%3 -> Q/K/V) into windowed Q and padded K/V.
// ---------------------------------------------------------------------------
__global__ void k_qkv(const _Float16* __restrict__ xt, const _Float16* __restrict__ w1h,
                      const float* __restrict__ b1, _Float16* __restrict__ Qw,
                      _Float16* __restrict__ Kp, _Float16* __restrict__ Vp) {
  __shared__ _Float16 Xs[64 * 34];
  __shared__ _Float16 Wsh[64 * 34];
  int m0 = blockIdx.x * 64, n0 = blockIdx.y * 64;
  int tid = threadIdx.x, lane = tid & 31, wave = tid >> 5;
  v8f acc[4];
#pragma unroll
  for (int j = 0; j < 4; ++j) acc[j] = zero8();

  for (int kt = 0; kt < 8; ++kt) {
    int k0 = kt * 32;
    for (int i = tid; i < 2048; i += 128) {
      int r = i >> 5, c = i & 31;
      Xs[r * 34 + c]  = xt[(size_t)(m0 + r) * 256 + k0 + c];
      Wsh[r * 34 + c] = w1h[(size_t)(n0 + r) * 256 + k0 + c];
    }
    __syncthreads();
    v16h a = load_a_frag(Xs, 34, wave * 16, 0, lane);
#pragma unroll
    for (int j = 0; j < 4; ++j) {
      v16h bb = load_b_frag_nk(Wsh, 34, j * 16, 0, lane);
      acc[j] = wmma_f16(a, bb, acc[j]);
    }
    __syncthreads();
  }

  int nl = lane & 15, half = lane >> 4;
#pragma unroll
  for (int j = 0; j < 4; ++j) {
#pragma unroll
    for (int r = 0; r < 8; ++r) {
      int col = n0 + j * 16 + nl;
      int m   = m0 + wave * 16 + half * 8 + r;
      float val = acc[j][r] + b1[col];
      int c = col / 3, rem = col - 3 * c;
      int head = c >> 5, e = c & 31;
      int b_ = m >> 12, hw = m & 4095, h = hw >> 6, w = hw & 63;
      _Float16 hv = (_Float16)val;
      if (rem == 0) {
        size_t qi = ((size_t)(((b_ * 8 + head) * 8 + (h >> 3)) * 8 + (w >> 3))) * 2048
                  + ((h & 7) * 8 + (w & 7)) * 32 + e;
        Qw[qi] = hv;
      } else {
        size_t pi = ((size_t)((b_ * 8 + head) * HP + h + 2) * HP + (w + 2)) * 32 + e;
        if (rem == 1) Kp[pi] = hv; else Vp[pi] = hv;
      }
    }
  }
}

// ---------------------------------------------------------------------------
// 3) Attention: one block per (b, head, wi, wj); 4 waves, each owns 16 queries.
//    scores = Q(64x32) @ K^T(32x144) -> softmax(144) -> @ V(160x32)
// ---------------------------------------------------------------------------
__global__ void k_attn(const _Float16* __restrict__ Qw, const _Float16* __restrict__ Kp,
                       const _Float16* __restrict__ Vp, _Float16* __restrict__ ao) {
  extern __shared__ char smem[];
  _Float16* Qs = (_Float16*)smem;            // 64  x 34
  _Float16* Ks = Qs + 64 * 34;               // 160 x 34
  _Float16* Vs = Ks + 160 * 34;              // 160 x 34
  float*    Sc = (float*)(Vs + 160 * 34);    // 4 x 16 x 168 f32
  _Float16* Aw = (_Float16*)(Sc + 4 * 16 * 168);  // 4 x 16 x 168 f16

  int idx = blockIdx.x;
  int wj = idx & 7, wi = (idx >> 3) & 7, head = (idx >> 6) & 7, b = idx >> 9;
  int tid = threadIdx.x, lane = tid & 31, wave = tid >> 5;
  int nl = lane & 15, half = lane >> 4;

  const _Float16* Qp = Qw + (size_t)(((b * 8 + head) * 8 + wi) * 8 + wj) * 2048;
  size_t kvbase = (size_t)(b * 8 + head) * HP * HP * 32;

#if USE_TDM
  // Tensor Data Mover: window gathers straight into the padded LDS layout.
  if (wave == 0) {
    const _Float16* Kg = Kp + kvbase + (size_t)((wi * 8) * HP + wj * 8) * 32;
    const _Float16* Vg = Vp + kvbase + (size_t)((wi * 8) * HP + wj * 8) * 32;
    // Q: 1D tile of 2048 f16 -> 64 rows x 34 in LDS via padding
    tdm_load_2d(lds_off(Qs), (unsigned long long)Qp, 2048, 1, 2048, 1, 2048);
    // K/V: 12 rows x 384 f16, row stride 68*32 -> 144 keys x 34 in LDS
    tdm_load_2d(lds_off(Ks), (unsigned long long)Kg, HP * 32, HP, MO * 32, MO, HP * 32);
    tdm_load_2d(lds_off(Vs), (unsigned long long)Vg, HP * 32, HP, MO * 32, MO, HP * 32);
    __builtin_amdgcn_s_wait_tensorcnt(0);
  }
  // zero the 16 padded key rows (keys 144..159)
  for (int i = tid; i < 16 * 34; i += 128) {
    Ks[NKEY * 34 + i] = (_Float16)0.f;
    Vs[NKEY * 34 + i] = (_Float16)0.f;
  }
#else
  for (int i = tid; i < 2048; i += 128) Qs[(i >> 5) * 34 + (i & 31)] = Qp[i];
  for (int i = tid; i < NKEY_PAD * 32; i += 128) {
    int kk = i >> 5, e = i & 31;
    _Float16 kv = (_Float16)0.f, vv = (_Float16)0.f;
    if (kk < NKEY) {
      int r = kk / MO, cc = kk - r * MO;
      size_t off = kvbase + ((size_t)((wi * 8 + r) * HP + wj * 8 + cc)) * 32 + e;
      kv = Kp[off];
      vv = Vp[off];
    }
    Ks[kk * 34 + e] = kv;
    Vs[kk * 34 + e] = vv;
  }
#endif
  __syncthreads();

  const float scale = 0.17677669529663687f;  // 1/sqrt(32)
  v16h aq = load_a_frag(Qs, 34, wave * 16, 0, lane);
  float* Scw = Sc + wave * 16 * 168;
#pragma unroll
  for (int nt = 0; nt < 9; ++nt) {            // 9*16 == 144 keys, exact
    v16h bk = load_b_frag_nk(Ks, 34, nt * 16, 0, lane);
    v8f c = wmma_f16(aq, bk, zero8());
#pragma unroll
    for (int r = 0; r < 8; ++r)
      Scw[(half * 8 + r) * 168 + nt * 16 + nl] = c[r] * scale;
  }
  __syncthreads();

  // Softmax over 144 keys: 2 lanes per query row, 72 columns each,
  // combined across halves with wave shuffles.
  _Float16* Aww = Aw + wave * 16 * 168;
  {
    float* row = Scw + nl * 168;
    _Float16* arow = Aww + nl * 168;
    int c0 = half * 72;
    float mx = -1e30f;
    for (int c = 0; c < 72; ++c) mx = fmaxf(mx, row[c0 + c]);
    mx = fmaxf(mx, __shfl_xor(mx, 16, 32));
    float s = 0.f;
    for (int c = 0; c < 72; ++c) {
      float e_ = __expf(row[c0 + c] - mx);
      row[c0 + c] = e_;
      s += e_;
    }
    s += __shfl_xor(s, 16, 32);
    float inv = 1.f / s;
    for (int c = 0; c < 72; ++c) arow[c0 + c] = (_Float16)(row[c0 + c] * inv);
#pragma unroll
    for (int i = 0; i < 8; ++i) arow[NKEY + half * 8 + i] = (_Float16)0.f;
  }
  __syncthreads();

#pragma unroll
  for (int et = 0; et < 2; ++et) {
    v8f c = zero8();
#pragma unroll
    for (int kt = 0; kt < 5; ++kt) {          // K = 160 = 5 * 32
      v16h aa = load_a_frag(Aww, 168, 0, kt * 32, lane);
      v16h bv = load_b_frag_kn(Vs, 34, kt * 32, et * 16, lane);
      c = wmma_f16(aa, bv, c);
    }
#pragma unroll
    for (int r = 0; r < 8; ++r) {
      int q = wave * 16 + half * 8 + r;
      int h = wi * 8 + (q >> 3), w = wj * 8 + (q & 7);
      int e = et * 16 + nl;
      ao[(size_t)(b * 4096 + h * 64 + w) * 256 + head * 32 + e] = (_Float16)c[r];
    }
  }
}

// ---------------------------------------------------------------------------
// 4) Output projection: y = ao @ W2^T + b2 -> (B,C,H,W) f32, coalesced stores
// ---------------------------------------------------------------------------
__global__ void k_proj(const _Float16* __restrict__ ao, const _Float16* __restrict__ w2h,
                       const float* __restrict__ b2, float* __restrict__ y) {
  __shared__ _Float16 Xs[64 * 34];
  __shared__ _Float16 Wsh[64 * 34];
  __shared__ float Ot[64 * 65];
  int m0 = blockIdx.x * 64, n0 = blockIdx.y * 64;
  int tid = threadIdx.x, lane = tid & 31, wave = tid >> 5;
  v8f acc[4];
#pragma unroll
  for (int j = 0; j < 4; ++j) acc[j] = zero8();

  for (int kt = 0; kt < 8; ++kt) {
    int k0 = kt * 32;
    for (int i = tid; i < 2048; i += 128) {
      int r = i >> 5, c = i & 31;
      Xs[r * 34 + c]  = ao[(size_t)(m0 + r) * 256 + k0 + c];
      Wsh[r * 34 + c] = w2h[(size_t)(n0 + r) * 256 + k0 + c];
    }
    __syncthreads();
    v16h a = load_a_frag(Xs, 34, wave * 16, 0, lane);
#pragma unroll
    for (int j = 0; j < 4; ++j) {
      v16h bb = load_b_frag_nk(Wsh, 34, j * 16, 0, lane);
      acc[j] = wmma_f16(a, bb, acc[j]);
    }
    __syncthreads();
  }

  int nl = lane & 15, half = lane >> 4;
#pragma unroll
  for (int j = 0; j < 4; ++j)
#pragma unroll
    for (int r = 0; r < 8; ++r)
      Ot[(wave * 16 + half * 8 + r) * 65 + j * 16 + nl] = acc[j][r];
  __syncthreads();

  int b_ = m0 >> 12, hw0 = m0 & 4095;
  for (int i = tid; i < 4096; i += 128) {
    int cc = i >> 6, tok = i & 63;
    float v = Ot[tok * 65 + cc] + b2[n0 + cc];
    y[(size_t)(b_ * 256 + n0 + cc) * 4096 + hw0 + tok] = v;
  }
}

// ---------------------------------------------------------------------------
// Workspace layout (bytes), total ~143.4 MB:
//   [0)         W1 f16          393216
//   [393216)    W2 f16          131072
//   [524288)    Qw f16        33554432
//   [34078720)  Kpad f16      37879808
//   [71958528)  Vpad f16      37879808
//   [109838336) xt / attn_out f16 (reused) 33554432
// ---------------------------------------------------------------------------
extern "C" void kernel_launch(void* const* d_in, const int* in_sizes, int n_in,
                              void* d_out, int out_size, void* d_ws, size_t ws_size,
                              hipStream_t stream) {
  const float* x  = (const float*)d_in[0];
  const float* W1 = (const float*)d_in[1];
  const float* b1 = (const float*)d_in[2];
  const float* W2 = (const float*)d_in[3];
  const float* b2 = (const float*)d_in[4];
  float* y = (float*)d_out;

  char* ws = (char*)d_ws;
  _Float16* w1h = (_Float16*)(ws + 0);
  _Float16* w2h = (_Float16*)(ws + 393216);
  _Float16* Qw  = (_Float16*)(ws + 524288);
  _Float16* Kp  = (_Float16*)(ws + 34078720);
  _Float16* Vp  = (_Float16*)(ws + 71958528);
  _Float16* xt  = (_Float16*)(ws + 109838336);  // reused as attn_out

  // Zero K/V padded images (2-pixel zero borders == reference's jnp.pad)
  hipMemsetAsync(Kp, 0, (size_t)2 * 37879808, stream);

  k_prep<<<1024, 256, 0, stream>>>(W1, W2, w1h, w2h);
  k_transpose<<<dim3(128, 8, 16), dim3(32, 8), 0, stream>>>(x, xt);
  k_qkv<<<dim3(1024, 12), 128, 0, stream>>>(xt, w1h, b1, Qw, Kp, Vp);

  size_t smem_attn = (size_t)(64 * 34 + 2 * 160 * 34) * sizeof(_Float16)
                   + (size_t)(4 * 16 * 168) * sizeof(float)
                   + (size_t)(4 * 16 * 168) * sizeof(_Float16);   // 90624 B
  k_attn<<<8192, 128, smem_attn, stream>>>(Qw, Kp, Vp, xt);

  k_proj<<<dim3(1024, 4), 128, 0, stream>>>(xt, w2h, b2, y);
}
